// CrossAttention_2250562863362
// MI455X (gfx1250) — matrix operand order repaired and checked by reading, not compiled, not run on hardware
//
#include <hip/hip_runtime.h>
#include <hip/hip_bf16.h>

typedef unsigned short u16;
typedef __bf16 bf16_t;
typedef bf16_t v16bf __attribute__((ext_vector_type(16)));
typedef bf16_t v8bf  __attribute__((ext_vector_type(8)));
typedef float  v8f   __attribute__((ext_vector_type(8)));

union V16 { v16bf v; v8bf h[2]; };

#define B_  4
#define S_  2048
#define D_  768
#define M_  (B_ * S_)          // 8192 rows total
#define NIMG ((size_t)M_ * D_) // 6291456
#define NW   ((size_t)D_ * D_) // 589824
#define NATT ((size_t)B_ * S_ * S_)

// LDS tile: 128 rows x 32 cols, row stride padded to 40 u16 (80 B = 20 dwords)
// -> 16-lane fragment reads hit 16 distinct banks; 16 B alignment preserved.
#define LDS_STRIDE 40

__device__ __forceinline__ float bf2f(u16 h) {
    return __uint_as_float(((unsigned int)h) << 16);
}
__device__ __forceinline__ u16 f2bf(float f) {
    unsigned int u = __float_as_uint(f);
    unsigned int r = (u + 0x7fffu + ((u >> 16) & 1u)) >> 16;  // RNE
    return (u16)r;
}

// Async global->LDS copy of 16 bytes (CDNA5 GLOBAL_LOAD_ASYNC_TO_LDS_B128,
// tracked by ASYNCcnt). LDS address = low 32 bits of the generic pointer
// (flat-aperture rule).
__device__ __forceinline__ void async_cp16(u16* l, const u16* g) {
    unsigned lds = (unsigned)(unsigned long long)l;
    unsigned long long ga = (unsigned long long)g;
    asm volatile("global_load_async_to_lds_b128 %0, %1, off"
                 :: "v"(lds), "v"(ga) : "memory");
}
__device__ __forceinline__ void async_wait0() {
    asm volatile("s_wait_asynccnt 0x0" ::: "memory");
}

// ---------------------------------------------------------------------------
// LayerNorm: one wave32 per 768-element row; 24 elems/lane; shuffle reduction.
// Rows [0, 3*M): tensor = row / M  (0=img, 1=ref_pose, 2=pose).
// ---------------------------------------------------------------------------
__global__ __launch_bounds__(256) void layernorm_bf16(
    const float* __restrict__ img, const float* __restrict__ ref,
    const float* __restrict__ pose, const float* __restrict__ gamma,
    const float* __restrict__ beta, u16* __restrict__ img_n,
    u16* __restrict__ ref_n, u16* __restrict__ pose_n) {
    const int wave = threadIdx.x >> 5;
    const int lane = threadIdx.x & 31;
    const int row  = blockIdx.x * 8 + wave;           // [0, 3*M)
    const int which = row / M_;
    const int r     = row - which * M_;

    const float* src = (which == 0) ? img : (which == 1) ? ref : pose;
    u16*         dst = (which == 0) ? img_n : (which == 1) ? ref_n : pose_n;
    const float* p = src + (size_t)r * D_;

    float x[24];
#pragma unroll
    for (int c = 0; c < 6; ++c) {
        float4 v = *reinterpret_cast<const float4*>(p + c * 128 + lane * 4);
        x[c * 4 + 0] = v.x; x[c * 4 + 1] = v.y;
        x[c * 4 + 2] = v.z; x[c * 4 + 3] = v.w;
    }
    float s = 0.f, sq = 0.f;
#pragma unroll
    for (int i = 0; i < 24; ++i) { s += x[i]; sq += x[i] * x[i]; }
#pragma unroll
    for (int off = 16; off >= 1; off >>= 1) {
        s  += __shfl_xor(s, off, 32);
        sq += __shfl_xor(sq, off, 32);
    }
    const float mean = s * (1.0f / D_);
    const float var  = sq * (1.0f / D_) - mean * mean;
    const float inv  = rsqrtf(var + 1e-5f);

    u16* drow = dst + (size_t)r * D_;
#pragma unroll
    for (int c = 0; c < 6; ++c) {
        const int col = c * 128 + lane * 4;
        float4 g = *reinterpret_cast<const float4*>(gamma + col);
        float4 bb = *reinterpret_cast<const float4*>(beta + col);
        ushort4 o;
        o.x = f2bf((x[c * 4 + 0] - mean) * inv * g.x + bb.x);
        o.y = f2bf((x[c * 4 + 1] - mean) * inv * g.y + bb.y);
        o.z = f2bf((x[c * 4 + 2] - mean) * inv * g.z + bb.z);
        o.w = f2bf((x[c * 4 + 3] - mean) * inv * g.w + bb.w);
        *reinterpret_cast<ushort4*>(drow + col) = o;
    }
}

// ---------------------------------------------------------------------------
// fp32 -> bf16 weight conversion (Wq, Wk, Wv, Wp)
// ---------------------------------------------------------------------------
__global__ __launch_bounds__(256) void conv_w(
    const float* __restrict__ wq, const float* __restrict__ wk,
    const float* __restrict__ wv, const float* __restrict__ wp,
    u16* __restrict__ oq, u16* __restrict__ ok2, u16* __restrict__ ov,
    u16* __restrict__ op, int nW) {
    int i = blockIdx.x * 256 + threadIdx.x;
    int which = i / nW;
    int r = i - which * nW;
    const float* s = (which == 0) ? wq : (which == 1) ? wk : (which == 2) ? wv : wp;
    u16* d = (which == 0) ? oq : (which == 1) ? ok2 : (which == 2) ? ov : op;
    d[r] = f2bf(s[r]);
}

// ---------------------------------------------------------------------------
// Fill one 128x32 LDS tile pair (A and B) with async global->LDS b128 copies.
// 128 threads: thread t handles row (t>>2)+32*it, 16-byte chunk (t&3).
// 8 async ops per thread per fill.
// ---------------------------------------------------------------------------
__device__ __forceinline__ void async_fill_tiles(
    u16* __restrict__ dstA, u16* __restrict__ dstB,
    const u16* __restrict__ gA, const u16* __restrict__ gB,
    int lda, int ldw, int tid) {
#pragma unroll
    for (int it = 0; it < 4; ++it) {
        const int row = (tid >> 2) + it * 32;
        const int chunk = (tid & 3) * 8;
        async_cp16(dstA + row * LDS_STRIDE + chunk, gA + (size_t)row * lda + chunk);
        async_cp16(dstB + row * LDS_STRIDE + chunk, gB + (size_t)row * ldw + chunk);
    }
    // Warm L2 for the tile after next (speculative, safe if OOB).
    __builtin_prefetch(gA + (size_t)(tid) * lda + 64, 0, 1);
    __builtin_prefetch(gB + (size_t)(tid) * ldw + 64, 0, 1);
}

// ---------------------------------------------------------------------------
// Generic bf16 WMMA GEMM:  C[z] = A[z] (MxK, row) x W[z]^T (W is NxK, row)
//   + bias (fp32, per-N, optional) + r1+r2 residuals (bf16, optional).
// MODE 0: store bf16 C[z*cBatch + gm*ldc + gn]
// MODE 1: store bf16 transposed-batched: b=gm/rowsPT, ml=gm-b*rowsPT,
//         C[b*ctBatch + gn*ldct + ml]
// MODE 2: store fp32 C
// Block = 128 threads (4 waves), 128x128 block tile, 64x64 wave tile =
// 4x4 grid of v_wmma_f32_16x16x32_bf16. K step 32, double-buffered async-LDS.
// ---------------------------------------------------------------------------
template <int MODE>
__global__ __launch_bounds__(128) void gemm_bf16_wmma(
    const u16* __restrict__ A, long long aBatch, int lda,
    const u16* __restrict__ Wm, long long wBatch, int ldw,
    const float* __restrict__ bias,
    const u16* __restrict__ r1, const u16* __restrict__ r2,
    long long rBatch, int ldr,
    u16* __restrict__ outB, float* __restrict__ outF,
    long long cBatch, int ldc,
    int Kdim, int rowsPT, long long ctBatch, int ldct) {
    __shared__ u16 sA[2][128 * LDS_STRIDE];
    __shared__ u16 sB[2][128 * LDS_STRIDE];

    const int tid  = threadIdx.x;
    const int lane = tid & 31;
    const int half = lane >> 4;       // 0: lanes 0-15, 1: lanes 16-31
    const int l16  = lane & 15;
    const int wave = tid >> 5;
    const int z    = blockIdx.z;
    const int m0blk = blockIdx.y * 128;
    const int n0blk = blockIdx.x * 128;
    const int wm64 = (wave >> 1) * 64;
    const int wn64 = (wave & 1) * 64;

    const u16* gA0 = A + (long long)z * aBatch + (size_t)m0blk * lda;
    const u16* gB0 = Wm + (long long)z * wBatch + (size_t)n0blk * ldw;

    v8f acc[4][4];
#pragma unroll
    for (int i = 0; i < 4; ++i)
#pragma unroll
        for (int j = 0; j < 4; ++j)
#pragma unroll
            for (int e = 0; e < 8; ++e) acc[i][j][e] = 0.f;

    async_fill_tiles(&sA[0][0], &sB[0][0], gA0, gB0, lda, ldw, tid);

    int p = 0;
    for (int kt = 0; kt < Kdim; kt += 32) {
        async_wait0();        // own async fills landed in LDS
        __syncthreads();      // everyone's fills landed; prior reads of buf p^1 done

        if (kt + 32 < Kdim)
            async_fill_tiles(&sA[p ^ 1][0], &sB[p ^ 1][0],
                             gA0 + kt + 32, gB0 + kt + 32, lda, ldw, tid);

        // Fragment loads from LDS per CDNA5 16-bit WMMA layouts.
        // A (16x32): lanes 0-15 K {0..7,16..23}; lanes 16-31 +8.
        // B (32x16): lane = column; lanes 0-15 K {0..7,16..23}; lanes 16-31 +8.
        V16 a[4], b[4];
#pragma unroll
        for (int i = 0; i < 4; ++i) {
            const u16* rp = &sA[p][(wm64 + i * 16 + l16) * LDS_STRIDE + half * 8];
            a[i].h[0] = *reinterpret_cast<const v8bf*>(rp);
            a[i].h[1] = *reinterpret_cast<const v8bf*>(rp + 16);
        }
#pragma unroll
        for (int j = 0; j < 4; ++j) {
            const u16* rp = &sB[p][(wn64 + j * 16 + l16) * LDS_STRIDE + half * 8];
            b[j].h[0] = *reinterpret_cast<const v8bf*>(rp);
            b[j].h[1] = *reinterpret_cast<const v8bf*>(rp + 16);
        }

#pragma unroll
        for (int i = 0; i < 4; ++i)
#pragma unroll
            for (int j = 0; j < 4; ++j)
                acc[i][j] = __builtin_amdgcn_wmma_f32_16x16x32_bf16(
                    false, a[i].v, false, b[j].v, (short)0, acc[i][j],
                    false, false);
        p ^= 1;
    }

    // Epilogue. C/D layout: VGPR r -> (M = r + half*8, N = l16) within tile.
    const int m0 = m0blk + wm64;
    const int n0 = n0blk + wn64;
#pragma unroll
    for (int i = 0; i < 4; ++i) {
#pragma unroll
        for (int j = 0; j < 4; ++j) {
            v8f c = acc[i][j];
            const int gn = n0 + j * 16 + l16;
            const float bb = bias ? bias[gn] : 0.f;
#pragma unroll
            for (int r = 0; r < 8; ++r) {
                const int gm = m0 + i * 16 + r + half * 8;
                float v = c[r] + bb;
                if (r1) {
                    const long long ri = (long long)z * rBatch + (long long)gm * ldr + gn;
                    v += bf2f(r1[ri]) + bf2f(r2[ri]);
                }
                if (MODE == 2) {
                    outF[(long long)z * cBatch + (long long)gm * ldc + gn] = v;
                } else if (MODE == 1) {
                    const int bidx = gm / rowsPT;
                    const int ml = gm - bidx * rowsPT;
                    outB[(long long)bidx * ctBatch + (long long)gn * ldct + ml] = f2bf(v);
                } else {
                    outB[(long long)z * cBatch + (long long)gm * ldc + gn] = f2bf(v);
                }
            }
        }
    }
}

// ---------------------------------------------------------------------------
// Row softmax over S=2048 bf16 logits (scale applied on load), in place.
// One 256-thread block per row; 8 elements per thread.
// ---------------------------------------------------------------------------
__global__ __launch_bounds__(256) void softmax_rows(u16* __restrict__ att,
                                                    float scale) {
    const int t = threadIdx.x;
    const int lane = t & 31;
    const int wave = t >> 5;
    const size_t base = (size_t)blockIdx.x * S_;

    float v[8];
#pragma unroll
    for (int k = 0; k < 8; ++k)
        v[k] = bf2f(att[base + k * 256 + t]) * scale;

    float m = v[0];
#pragma unroll
    for (int k = 1; k < 8; ++k) m = fmaxf(m, v[k]);
#pragma unroll
    for (int off = 16; off >= 1; off >>= 1) m = fmaxf(m, __shfl_xor(m, off, 32));

    __shared__ float sm[8];
    __shared__ float ss[8];
    if (lane == 0) sm[wave] = m;
    __syncthreads();
    float bm = sm[0];
#pragma unroll
    for (int w = 1; w < 8; ++w) bm = fmaxf(bm, sm[w]);

    float s = 0.f;
#pragma unroll
    for (int k = 0; k < 8; ++k) { v[k] = __expf(v[k] - bm); s += v[k]; }
#pragma unroll
    for (int off = 16; off >= 1; off >>= 1) s += __shfl_xor(s, off, 32);
    if (lane == 0) ss[wave] = s;
    __syncthreads();
    float bs = 0.f;
#pragma unroll
    for (int w = 0; w < 8; ++w) bs += ss[w];

    const float inv = 1.f / bs;
#pragma unroll
    for (int k = 0; k < 8; ++k)
        att[base + k * 256 + t] = f2bf(v[k] * inv);
}

// ---------------------------------------------------------------------------
extern "C" void kernel_launch(void* const* d_in, const int* in_sizes, int n_in,
                              void* d_out, int out_size, void* d_ws,
                              size_t ws_size, hipStream_t stream) {
    (void)in_sizes; (void)n_in; (void)out_size; (void)ws_size;

    const float* img   = (const float*)d_in[0];
    const float* refp  = (const float*)d_in[1];
    const float* pose  = (const float*)d_in[2];
    const float* Wq    = (const float*)d_in[3];
    const float* bq    = (const float*)d_in[4];
    const float* Wk    = (const float*)d_in[5];
    const float* bk    = (const float*)d_in[6];
    const float* Wv    = (const float*)d_in[7];
    const float* bv    = (const float*)d_in[8];
    const float* Wp    = (const float*)d_in[9];
    const float* bp    = (const float*)d_in[10];
    const float* gamma = (const float*)d_in[11];
    const float* beta  = (const float*)d_in[12];
    float* out = (float*)d_out;

    // Workspace carve-up (bf16 as u16). All sizes multiple of 128 elements.
    u16* p = (u16*)d_ws;
    u16* img_n = p; p += NIMG;
    u16* ref_n = p; p += NIMG;
    u16* pose_n = p; p += NIMG;
    u16* WqB = p; p += NW;
    u16* WkB = p; p += NW;
    u16* WvB = p; p += NW;
    u16* WpB = p; p += NW;
    u16* Qb = p; p += NIMG;
    u16* Kb = p; p += NIMG;
    u16* Vt = p; p += NIMG;          // [B][D][S]  (V transposed per batch)
    u16* att = p; p += NATT;         // [B][S][S]
    u16* H = Qb;                     // reuse: Q dead after QK^T

    // 1) LayerNorm all three inputs -> bf16
    layernorm_bf16<<<dim3(3 * M_ / 8), dim3(256), 0, stream>>>(
        img, refp, pose, gamma, beta, img_n, ref_n, pose_n);

    // 2) Weights -> bf16
    conv_w<<<dim3(4 * (int)NW / 256), dim3(256), 0, stream>>>(
        Wq, Wk, Wv, Wp, WqB, WkB, WvB, WpB, (int)NW);

    // 3) Q = ref_n Wq^T + bq ; K = pose_n Wk^T + bk ; V^T = (img_n Wv^T + bv)^T
    gemm_bf16_wmma<0><<<dim3(D_ / 128, M_ / 128, 1), dim3(128), 0, stream>>>(
        ref_n, 0, D_, WqB, 0, D_, bq, nullptr, nullptr, 0, 0,
        Qb, nullptr, 0, D_, D_, 0, 0, 0);
    gemm_bf16_wmma<0><<<dim3(D_ / 128, M_ / 128, 1), dim3(128), 0, stream>>>(
        pose_n, 0, D_, WkB, 0, D_, bk, nullptr, nullptr, 0, 0,
        Kb, nullptr, 0, D_, D_, 0, 0, 0);
    gemm_bf16_wmma<1><<<dim3(D_ / 128, M_ / 128, 1), dim3(128), 0, stream>>>(
        img_n, 0, D_, WvB, 0, D_, bv, nullptr, nullptr, 0, 0,
        Vt, nullptr, 0, D_, D_, /*rowsPT*/S_,
        /*ctBatch*/(long long)D_ * S_, /*ldct*/S_);

    // 4) att[b] = Q[b] K[b]^T   (batched over grid.z)
    gemm_bf16_wmma<0><<<dim3(S_ / 128, S_ / 128, B_), dim3(128), 0, stream>>>(
        Qb, (long long)S_ * D_, D_, Kb, (long long)S_ * D_, D_, nullptr,
        nullptr, nullptr, 0, 0,
        att, nullptr, (long long)S_ * S_, S_, D_, 0, 0, 0);

    // 5) softmax rows (scale = D^-0.5 on load)
    softmax_rows<<<dim3(M_), dim3(256), 0, stream>>>(att, 0.03608439182435161f);

    // 6) H[b] = att[b] Vt[b]^T + img_n + pose_n   (Vt is [D][S] => W NxK form)
    gemm_bf16_wmma<0><<<dim3(D_ / 128, S_ / 128, B_), dim3(128), 0, stream>>>(
        att, (long long)S_ * S_, S_, Vt, (long long)D_ * S_, S_, nullptr,
        img_n, pose_n, (long long)S_ * D_, D_,
        H, nullptr, (long long)S_ * D_, D_, S_, 0, 0, 0);

    // 7) out = H Wp^T + bp  (fp32 store to d_out)
    gemm_bf16_wmma<2><<<dim3(D_ / 128, M_ / 128, 1), dim3(128), 0, stream>>>(
        H, 0, D_, WpB, 0, D_, bp, nullptr, nullptr, 0, 0,
        nullptr, out, 0, D_, D_, 0, 0, 0);
}
